// PILayer_74818330296694
// MI455X (gfx1250) — matrix operand order repaired and checked by reading, not compiled, not run on hardware
//
#include <hip/hip_runtime.h>
#include <hip/hip_bf16.h>

// Problem constants (from reference)
#define P_      256
#define T_      65536
#define UNITS_  128
// Scan chunking: NC_ chunks of CHUNK_ steps each (NC_*CHUNK_ == T_)
#define NC_     256
#define CHUNK_  256
// GEMM split-K
#define KSLICE_ 1024
#define KS_     (T_ / KSLICE_)   // 64 K-slices

typedef __attribute__((ext_vector_type(16))) __bf16 bf16x16;
typedef __attribute__((ext_vector_type(8)))  __bf16 bf16x8;
typedef __attribute__((ext_vector_type(8)))  float  f32x8;

// ---- workspace layout (bytes) ----
static constexpr size_t OUTS_OFF  = 0;                                   // P*T bf16
static constexpr size_t OUTS_SZ   = (size_t)P_ * T_ * 2;                 // 32 MB
static constexpr size_t KT_OFF    = OUTS_OFF + OUTS_SZ;                  // UNITS*T bf16 (N-major)
static constexpr size_t KT_SZ     = (size_t)UNITS_ * T_ * 2;             // 16 MB
static constexpr size_t LO_OFF    = KT_OFF + KT_SZ;                      // P*NC f32
static constexpr size_t LO_SZ     = (size_t)P_ * NC_ * 4;
static constexpr size_t HI_OFF    = LO_OFF + LO_SZ;
static constexpr size_t HI_SZ     = LO_SZ;
static constexpr size_t ENT_OFF   = HI_OFF + HI_SZ;                      // P*NC f32 entry states
static constexpr size_t ENT_SZ    = LO_SZ;
static constexpr size_t ACC_OFF   = ENT_OFF + ENT_SZ;                    // P*UNITS f32 accum
static constexpr size_t ACC_SZ    = (size_t)P_ * UNITS_ * 4;

__device__ __forceinline__ unsigned short f2bf(float f) {
  unsigned u = __float_as_uint(f);
  if ((u & 0x7fffffffu) > 0x7f800000u) return (unsigned short)0x7fc0;  // canonical NaN
  u += 0x7fffu + ((u >> 16) & 1u);                                     // round-nearest-even
  return (unsigned short)(u >> 16);
}

// k1: convert kernel [T][UNITS] f32 -> kernelT [UNITS][T] bf16 (N-major so B
// fragments become contiguous K runs). Writes coalesced; reads via L2.
__global__ __launch_bounds__(256) void pi_convert_kernel(
    const float* __restrict__ kern, unsigned short* __restrict__ kernelT) {
  int gid = blockIdx.x * 256 + threadIdx.x;      // gid == n*T + t
  int n = gid >> 16;
  int t = gid & (T_ - 1);
  kernelT[(size_t)n * T_ + t] = f2bf(kern[(size_t)t * UNITS_ + n]);
}

// k2: per-(player,chunk) compose the 256-step clamp maps into (lo, hi):
//   s_out = min(max(s_in, lo), hi).  Per step (a=x, b=x+1):
//   lo = max(lo, a);  hi = min(max(hi, a), b)
__global__ __launch_bounds__(256) void pi_scan_compose(
    const float* __restrict__ in, const float* __restrict__ w,
    float* __restrict__ loOut, float* __restrict__ hiOut) {
  int gid = blockIdx.x * 256 + threadIdx.x;      // 0 .. P_*NC_-1
  int p = gid >> 8;                               // NC_ == 256
  int c = gid & (NC_ - 1);
  float wp = w[p];
  const float4* src = (const float4*)(in + (size_t)p * T_ + (size_t)c * CHUNK_);
  float lo = -3.4e38f, hi = 3.4e38f;
#pragma unroll 4
  for (int i = 0; i < CHUNK_ / 4; ++i) {
    float4 v = src[i];
    float x;
    x = v.x * wp; lo = fmaxf(lo, x); hi = fminf(fmaxf(hi, x), x + 1.0f);
    x = v.y * wp; lo = fmaxf(lo, x); hi = fminf(fmaxf(hi, x), x + 1.0f);
    x = v.z * wp; lo = fmaxf(lo, x); hi = fminf(fmaxf(hi, x), x + 1.0f);
    x = v.w * wp; lo = fmaxf(lo, x); hi = fminf(fmaxf(hi, x), x + 1.0f);
  }
  loOut[gid] = lo;
  hiOut[gid] = hi;
}

// k3: per-player serial carry across the 256 chunk maps (tiny: 256 thr x 256 it)
__global__ __launch_bounds__(256) void pi_scan_carry(
    const float* __restrict__ lo, const float* __restrict__ hi,
    const float* __restrict__ state, float* __restrict__ entry) {
  int p = threadIdx.x;
  float s = state[p];
  for (int c = 0; c < NC_; ++c) {
    int idx = p * NC_ + c;
    entry[idx] = s;
    s = fminf(fmaxf(s, lo[idx]), hi[idx]);
  }
}

// k4: replay each chunk from its known entry state, emit bf16 outs [P][T].
// Inputs are L2-resident after k2 (64 MB < 192 MB L2).
__global__ __launch_bounds__(256) void pi_scan_replay(
    const float* __restrict__ in, const float* __restrict__ w,
    const float* __restrict__ entry, unsigned short* __restrict__ outsBF) {
  int gid = blockIdx.x * 256 + threadIdx.x;
  int p = gid >> 8;
  int c = gid & (NC_ - 1);
  float wp = w[p];
  float s = entry[gid];
  size_t base = (size_t)p * T_ + (size_t)c * CHUNK_;
  const float4* src = (const float4*)(in + base);
  ushort4* dst = (ushort4*)(outsBF + base);
#pragma unroll 4
  for (int i = 0; i < CHUNK_ / 4; ++i) {
    float4 v = src[i];
    ushort4 o;
    float x;
    x = v.x * wp; s = fminf(fmaxf(s, x), x + 1.0f); o.x = f2bf(s);
    x = v.y * wp; s = fminf(fmaxf(s, x), x + 1.0f); o.y = f2bf(s);
    x = v.z * wp; s = fminf(fmaxf(s, x), x + 1.0f); o.z = f2bf(s);
    x = v.w * wp; s = fminf(fmaxf(s, x), x + 1.0f); o.w = f2bf(s);
    dst[i] = o;
  }
}

// k5: zero the f32 split-K accumulator (kernel, not memset: graph-capture safe)
__global__ __launch_bounds__(256) void pi_zero_acc(float* __restrict__ acc) {
  acc[blockIdx.x * 256 + threadIdx.x] = 0.0f;
}

// k6: split-K bf16 WMMA GEMM: outs[256 x 65536] @ kernelT^T -> acc[256 x 128].
// Block = 8 waves; wave w owns N-tile w; block -> (M-tile, K-slice).
// Fragment layouts (A and B share contiguous K = (L>>4)*16 + elem; C/D per CDNA5 ISA 7.12.2):
//   A 16x32 bf16: lane L -> M = L&15; elems0-15 = K[(L>>4)*16 .. +15] (two 16B loads)
//   B 32x16 bf16: lane L -> N = L&15; elems0-15 = K[(L>>4)*16 .. +15] (contiguous
//                 in the N-major kernelT copy -> one 32B load)
//   C/D f32: vgpr r, lane L -> M = r + (L>>4)*8, N = L&15
__global__ __launch_bounds__(256) void pi_gemm_wmma(
    const unsigned short* __restrict__ outsBF,   // [P][T] bf16
    const unsigned short* __restrict__ kernelT,  // [UNITS][T] bf16
    float* __restrict__ acc) {                   // [P][UNITS] f32
  const int lane  = threadIdx.x & 31;
  const int wave  = threadIdx.x >> 5;   // N-tile 0..7
  const int mtile = blockIdx.x;         // 0..15
  const int kslc  = blockIdx.y;         // 0..KS_-1

  const int m0 = mtile * 16;
  const int n0 = wave * 16;
  const int half = lane >> 4;
  const int lid  = lane & 15;
  const size_t kbase = (size_t)kslc * KSLICE_;

  const unsigned short* aPtr = outsBF  + (size_t)(m0 + lid) * T_ + kbase + half * 16;
  const unsigned short* bPtr = kernelT + (size_t)(n0 + lid) * T_ + kbase + half * 16;

  f32x8 c = {};
  for (int kk = 0; kk < KSLICE_; kk += 32) {
    __builtin_prefetch(aPtr + kk + 64, 0, 0);   // global_prefetch_b8
    __builtin_prefetch(bPtr + kk + 64, 0, 0);
    bf16x8  a0 = *(const bf16x8*)(aPtr + kk);        // K[half*16 .. +7]
    bf16x8  a1 = *(const bf16x8*)(aPtr + kk + 8);    // K[half*16+8 .. +7]
    bf16x16 b  = *(const bf16x16*)(bPtr + kk);       // 16 contiguous K
    bf16x16 a  = __builtin_shufflevector(a0, a1,
        0, 1, 2, 3, 4, 5, 6, 7, 8, 9, 10, 11, 12, 13, 14, 15);
    c = __builtin_amdgcn_wmma_f32_16x16x32_bf16(
        /*neg_a=*/false, a, /*neg_b=*/false, b,
        /*c_mod=*/(short)0, c, /*reuse_a=*/false, /*reuse_b=*/false);
  }

#pragma unroll
  for (int r = 0; r < 8; ++r) {
    int mOut = m0 + r + half * 8;
    atomicAdd(&acc[(size_t)mOut * UNITS_ + n0 + lid], c[r]);
  }
}

// k7: out = tanh(acc + bias)
__global__ __launch_bounds__(256) void pi_finalize(
    const float* __restrict__ acc, const float* __restrict__ bias,
    float* __restrict__ out) {
  int gid = blockIdx.x * 256 + threadIdx.x;    // 0 .. P_*UNITS_-1
  int n = gid & (UNITS_ - 1);
  out[gid] = tanhf(acc[gid] + bias[n]);
}

extern "C" void kernel_launch(void* const* d_in, const int* in_sizes, int n_in,
                              void* d_out, int out_size, void* d_ws, size_t ws_size,
                              hipStream_t stream) {
  const float* inputs = (const float*)d_in[0];   // [P,T]
  const float* pw     = (const float*)d_in[1];   // [P]
  const float* kern   = (const float*)d_in[2];   // [T,UNITS]
  const float* bias   = (const float*)d_in[3];   // [UNITS]
  const float* state  = (const float*)d_in[4];   // [P]
  float* out = (float*)d_out;                    // [P,UNITS]

  char* ws = (char*)d_ws;
  unsigned short* outsBF  = (unsigned short*)(ws + OUTS_OFF);
  unsigned short* kernelT = (unsigned short*)(ws + KT_OFF);
  float* lo    = (float*)(ws + LO_OFF);
  float* hi    = (float*)(ws + HI_OFF);
  float* entry = (float*)(ws + ENT_OFF);
  float* acc   = (float*)(ws + ACC_OFF);

  // Phase 1: kernel f32 -> bf16, N-major transpose (for contiguous B fragments)
  pi_convert_kernel<<<(T_ * UNITS_) / 256, 256, 0, stream>>>(kern, kernelT);
  // Phase 2: parallel play-operator scan via clamp-map composition
  pi_scan_compose<<<(P_ * NC_) / 256, 256, 0, stream>>>(inputs, pw, lo, hi);
  pi_scan_carry<<<1, P_, 0, stream>>>(lo, hi, state, entry);
  pi_scan_replay<<<(P_ * NC_) / 256, 256, 0, stream>>>(inputs, pw, entry, outsBF);
  // Phase 3: split-K WMMA GEMM with f32 atomic accumulation
  pi_zero_acc<<<(P_ * UNITS_) / 256, 256, 0, stream>>>(acc);
  pi_gemm_wmma<<<dim3(P_ / 16, KS_), 256, 0, stream>>>(outsBF, kernelT, acc);
  // Phase 4: bias + tanh
  pi_finalize<<<(P_ * UNITS_) / 256, 256, 0, stream>>>(acc, bias, out);
}
